// GVPAttentionConvLayer_19567871001004
// MI455X (gfx1250) — matrix operand order repaired and checked by reading, not compiled, not run on hardware
//
#include <hip/hip_runtime.h>
#include <hip/hip_bf16.h>

// ---------------------------------------------------------------------------
// GVP attention conv layer for MI455X (gfx1250, wave32, WMMA).
// Dense GVP linears run on v_wmma_f32_16x16x32_f16 (f16 in, f32 acc).
// Weights are pre-swizzled once into f16 WMMA-B fragment order (L2-resident),
// so each B fragment is one aligned 32B global load; the LDS A matrix is
// staged directly in fragment order, so each A fragment is one 32B DS load.
// Sigmoid/softmax normalizations use v_rcp_f32 / v_rsq_f32 transcendentals
// instead of IEEE division sequences.
// Segment softmax / aggregation via L2 float atomics (working set << 192MB L2).
// ---------------------------------------------------------------------------

#define NS 128
#define NV 16
#define ES 32
#define HEADS 4
#define HS 32
#define HV 4

typedef __attribute__((ext_vector_type(16))) _Float16 v16h;
typedef __attribute__((ext_vector_type(8)))  float    v8f;

// Fragment layouts (CDNA5 ISA 7.12.2), linearized so element index i is
// contiguous per lane:
//   A (16x32 f16): lane = m + 16*((kk>>3)&1), elem e = ((kk&16)>>1)|(kk&7), kk=k&31
//   B (32x16 f16): lane: n = (lane&15), h = lane>>4; elem i: k = 32t+((i&8)<<1)+8h+(i&7)
// Storage: frag_buf[ (tileStepIdx*32 + lane)*16 + e ]  -> 32B per lane per frag.

__device__ inline int afrag_off(int m, int c) {
  int t = c >> 5, kk = c & 31;
  int lane = m + (((kk >> 3) & 1) << 4);
  int e = ((kk & 16) >> 1) | (kk & 7);
  return ((t * 32 + lane) << 4) + e;
}

__device__ inline v16h load_frag_lds(const _Float16* base, int t) {
  int lane = threadIdx.x & 31;
  return *(const v16h*)(base + (((t * 32) + lane) << 4));
}

__device__ inline v16h load_frag_g(const _Float16* __restrict__ buf, int idx) {
  int lane = threadIdx.x & 31;
  return *(const v16h*)(buf + (((size_t)idx * 32 + lane) << 4));
}

__device__ inline float fast_rcp(float x) { return __builtin_amdgcn_rcpf(x); }
__device__ inline float sigmoidf_(float x) { return fast_rcp(1.f + __expf(-x)); }

// ----------------------------- weight pre-swizzle ---------------------------
// W [K,N] f32 row-major  ->  f16 B-fragment order, zero-padded K to mult of 32.
__global__ __launch_bounds__(256) void pack_w_kernel(const float* __restrict__ W,
                                                     _Float16* __restrict__ out,
                                                     int K, int N, long long total) {
  long long idx = (long long)blockIdx.x * blockDim.x + threadIdx.x;
  if (idx >= total) return;
  int steps = (((K + 31) & ~31) >> 5);
  int i    = (int)(idx & 15);
  int lane = (int)((idx >> 4) & 31);
  long long ts = idx >> 9;
  int t = (int)(ts % steps);
  int j = (int)(ts / steps);
  int k = t * 32 + ((i & 8) << 1) + ((lane >> 4) << 3) + (i & 7);
  int n = j * 16 + (lane & 15);
  out[idx] = (k < K) ? (_Float16)W[(size_t)k * N + n] : (_Float16)0.f;
}

// ----------------------------- generic fused GVP kernel ---------------------
// One block = 16 rows, 256 threads = 8 wave32 waves.
// s_out = silu?( [s | ||v @ wh||] @ ws_w + b )
// v_out = (vh @ wv)^T * sigmoid( g(s_out) @ wsv_w + b ),  g = sigmoid if VACT
template <int SI, int VI, int SO, int VO, int H, bool SACT, bool VACT>
__global__ __launch_bounds__(256) void gvp_kernel(
    const float* __restrict__ xs,        // [R, SI]
    const float* __restrict__ xv,        // [R, VI, 3]
    const float* __restrict__ wh,        // [VI, H]
    const _Float16* __restrict__ wsp,    // packed ws_w fragments
    const float* __restrict__ ws_b,      // [SO]
    const float* __restrict__ wv,        // [H, VO]
    const _Float16* __restrict__ wsvp,   // packed wsv_w fragments
    const float* __restrict__ wsv_b,     // [VO]
    float* __restrict__ os,              // [R, SO]
    float* __restrict__ ov,              // [R, VO, 3]
    int R) {
  constexpr int K     = SI + H;
  constexpr int KP    = (K + 31) & ~31;
  constexpr int STEPS = KP / 32;     // k steps of big GEMM
  constexpr int NT    = SO / 16;     // output N tiles
  constexpr int GST   = SO / 32;     // k steps of gate GEMM
  constexpr int GT    = VO / 16;     // gate N tiles

  __shared__ __align__(32) _Float16 sA[16 * KP];     // A fragments: [s | vn]
  __shared__ float                  sVH[16 * 3 * H]; // vh = v^T @ wh
  __shared__ float                  sOUT[16 * SO];   // f32 s_out
  __shared__ __align__(32) _Float16 sG[16 * SO];     // gate-input fragments
  __shared__ float                  sGate[16 * (VO > 0 ? VO : 1)];

  const int tid  = threadIdx.x;
  const int row0 = blockIdx.x * 16;

  // stage scalar features as f16 fragments (zero pad rows >= R)
  for (int idx = tid; idx < 16 * SI; idx += 256) {
    int r = idx / SI, c = idx - r * SI;
    int row = row0 + r;
    float v = (row < R) ? xs[(size_t)row * SI + c] : 0.f;
    sA[afrag_off(r, c)] = (_Float16)v;
  }
  // vh[r][d][h] = sum_k v[r][k][d] * wh[k][h]
  for (int idx = tid; idx < 16 * 3 * H; idx += 256) {
    int r = idx / (3 * H);
    int rem = idx - r * 3 * H;
    int d = rem / H, hh = rem - d * H;
    int row = row0 + r;
    float acc = 0.f;
    if (row < R) {
      for (int k = 0; k < VI; ++k)
        acc += xv[(size_t)row * VI * 3 + k * 3 + d] * wh[k * H + hh];
    }
    sVH[(r * 3 + d) * H + hh] = acc;
  }
  __syncthreads();
  // vn into columns SI..SI+H-1; zero-fill K..KP pad
  for (int idx = tid; idx < 16 * (KP - SI); idx += 256) {
    int r = idx / (KP - SI), c = idx - r * (KP - SI);
    float val = 0.f;
    if (c < H) {
      float a0 = sVH[(r * 3 + 0) * H + c];
      float a1 = sVH[(r * 3 + 1) * H + c];
      float a2 = sVH[(r * 3 + 2) * H + c];
      val = sqrtf(fmaxf(a0 * a0 + a1 * a1 + a2 * a2, 1e-8f));
    }
    sA[afrag_off(r, SI + c)] = (_Float16)val;
  }
  __syncthreads();

  const int wave = tid >> 5;
  // big GEMM: s_out = A[16,KP] @ ws_w[K,SO]  (wave-parallel over N tiles)
  for (int j = wave; j < NT; j += 8) {
    v8f acc = {};
#pragma unroll
    for (int t = 0; t < STEPS; ++t) {
      v16h a = load_frag_lds(sA, t);
      v16h b = load_frag_g(wsp, j * STEPS + t);
      acc = __builtin_amdgcn_wmma_f32_16x16x32_f16(false, a, false, b,
                                                   (short)0, acc, false, false);
    }
    int lane = tid & 31;
    int n    = j * 16 + (lane & 15);
    int mofs = (lane < 16) ? 0 : 8;   // C/D layout: VGPR r -> M = r (+8 hi lanes)
    float bias = ws_b[n];
#pragma unroll
    for (int r = 0; r < 8; ++r)
      sOUT[(r + mofs) * SO + n] = acc[r] + bias;
  }
  __syncthreads();

  if (VO > 0) {
    // gate input fragments (f16)
    for (int idx = tid; idx < 16 * SO; idx += 256) {
      int r = idx / SO, c = idx - r * SO;
      float v = sOUT[idx];
      sG[afrag_off(r, c)] = (_Float16)(VACT ? sigmoidf_(v) : v);
    }
    __syncthreads();
    // gate GEMM: [16,SO] @ wsv_w[SO,VO]
    if (wave < GT) {
      v8f acc = {};
#pragma unroll
      for (int t = 0; t < GST; ++t) {
        v16h a = load_frag_lds(sG, t);
        v16h b = load_frag_g(wsvp, wave * GST + t);
        acc = __builtin_amdgcn_wmma_f32_16x16x32_f16(false, a, false, b,
                                                     (short)0, acc, false, false);
      }
      int lane = tid & 31;
      int n    = wave * 16 + (lane & 15);
      int mofs = (lane < 16) ? 0 : 8;
      float bias = wsv_b[n];
#pragma unroll
      for (int r = 0; r < 8; ++r)
        sGate[(r + mofs) * VO + n] = sigmoidf_(acc[r] + bias);
    }
    __syncthreads();
    // v_out[r][o][d] = (sum_h vh[r][d][h] * wv[h][o]) * gate[r][o]
    for (int idx = tid; idx < 16 * VO * 3; idx += 256) {
      int r = idx / (VO * 3);
      int rem = idx - r * VO * 3;
      int o = rem / 3, d = rem - o * 3;
      int row = row0 + r;
      if (row < R) {
        float acc = 0.f;
        for (int hh = 0; hh < H; ++hh)
          acc += sVH[(r * 3 + d) * H + hh] * wv[hh * VO + o];
        ov[(size_t)row * VO * 3 + o * 3 + d] = acc * sGate[r * VO + o];
      }
    }
  }
  // s_out (+silu)
  for (int idx = tid; idx < 16 * SO; idx += 256) {
    int r = idx / SO, c = idx - r * SO;
    int row = row0 + r;
    if (row < R) {
      float v = sOUT[idx];
      if (SACT) v = v * sigmoidf_(v);
      os[(size_t)row * SO + c] = v;
    }
  }
}

// ----------------------------- segment softmax helpers ----------------------
__device__ inline unsigned f2key(float f) {
  unsigned u = __float_as_uint(f);
  return (u & 0x80000000u) ? ~u : (u | 0x80000000u);
}
__device__ inline float key2f(unsigned k) {
  unsigned u = (k & 0x80000000u) ? (k & 0x7FFFFFFFu) : ~k;
  return __uint_as_float(u);
}

__global__ void zero_kernel(float* __restrict__ p, long long n) {
  long long i = (long long)blockIdx.x * blockDim.x + threadIdx.x;
  if (i < n) p[i] = 0.f;
}

// attn-GVP energy per (edge, head) + segment max (keyed uint atomicMax)
__global__ __launch_bounds__(256) void energy_kernel(
    const int* __restrict__ ei, long long E,
    const float* __restrict__ q_s, const float* __restrict__ q_v,
    const float* __restrict__ k_s, const float* __restrict__ k_v,
    const float* __restrict__ ek_s, const float* __restrict__ ek_v,
    const float* __restrict__ awh,   // [8,8]
    const float* __restrict__ aws_w, // [72,1]
    const float* __restrict__ aws_b, // [1]
    float* __restrict__ energy, unsigned* __restrict__ maxbuf) {
  long long tid = (long long)blockIdx.x * blockDim.x + threadIdx.x;
  if (tid >= E * 4) return;
  int edge = (int)(tid >> 2);
  int h    = (int)(tid & 3);
  int src = ei[edge], dst = ei[E + edge];

  float vh[3][8];
#pragma unroll
  for (int d = 0; d < 3; ++d)
#pragma unroll
    for (int j = 0; j < 8; ++j) vh[d][j] = 0.f;

#pragma unroll
  for (int k = 0; k < 8; ++k) {   // a_v = [qv(4) | kv+ekv(4)]
    float vx[3];
    if (k < 4) {
      const float* p = q_v + (size_t)dst * 48 + (h * 4 + k) * 3;
      vx[0] = p[0]; vx[1] = p[1]; vx[2] = p[2];
    } else {
      int kk = k - 4;
      const float* p1 = k_v + (size_t)src * 48 + (h * 4 + kk) * 3;
      const float* p2 = ek_v + (size_t)edge * 48 + (h * 4 + kk) * 3;
      vx[0] = p1[0] + p2[0]; vx[1] = p1[1] + p2[1]; vx[2] = p1[2] + p2[2];
    }
#pragma unroll
    for (int j = 0; j < 8; ++j) {
      float w = awh[k * 8 + j];
      vh[0][j] += vx[0] * w; vh[1][j] += vx[1] * w; vh[2][j] += vx[2] * w;
    }
  }
  float e = aws_b[0];
#pragma unroll
  for (int j = 0; j < 8; ++j) {
    float s2 = vh[0][j] * vh[0][j] + vh[1][j] * vh[1][j] + vh[2][j] * vh[2][j];
    e += sqrtf(fmaxf(s2, 1e-8f)) * aws_w[64 + j];
  }
  const float* qsp  = q_s + (size_t)dst * NS + h * HS;
  const float* ksp  = k_s + (size_t)src * NS + h * HS;
  const float* eksp = ek_s + (size_t)edge * NS + h * HS;
  for (int i = 0; i < HS; ++i) {
    e += qsp[i] * aws_w[i];
    e += (ksp[i] + eksp[i]) * aws_w[HS + i];
  }
  energy[tid] = e;
  atomicMax(&maxbuf[(size_t)dst * 4 + h], f2key(e));
}

__global__ __launch_bounds__(256) void softmax_kernel(
    const int* __restrict__ ei, long long E, float* __restrict__ energy,
    const unsigned* __restrict__ maxbuf, float* __restrict__ den) {
  long long tid = (long long)blockIdx.x * blockDim.x + threadIdx.x;
  if (tid >= E * 4) return;
  int edge = (int)(tid >> 2);
  int h    = (int)(tid & 3);
  int dst  = ei[E + edge];
  float m  = key2f(maxbuf[(size_t)dst * 4 + h]);
  float ex = __expf(energy[tid] - m);
  energy[tid] = ex;                       // in place: now holds exp(e - max)
  atomicAdd(&den[(size_t)dst * 4 + h], ex);
}

// per (edge, channel) alpha-weighted atomic scatter: 128 scalar + 48 vector chans
__global__ __launch_bounds__(256) void agg_kernel(
    const int* __restrict__ ei, long long E,
    const float* __restrict__ ex, const float* __restrict__ den,
    const float* __restrict__ v_s, const float* __restrict__ v_v,
    const float* __restrict__ ev_s, const float* __restrict__ ev_v,
    float* __restrict__ agg_s, float* __restrict__ agg_v) {
  long long tid = (long long)blockIdx.x * blockDim.x + threadIdx.x;
  long long total = E * 176;
  if (tid >= total) return;
  int edge = (int)(tid / 176);
  int c    = (int)(tid - (long long)edge * 176);
  int src = ei[edge], dst = ei[E + edge];
  int h = (c < NS) ? (c >> 5) : ((c - NS) / (HV * 3));
  float alpha = ex[(size_t)edge * 4 + h] * fast_rcp(den[(size_t)dst * 4 + h] + 1e-16f);
  if (c < NS) {
    float val = v_s[(size_t)src * NS + c] + ev_s[(size_t)edge * NS + c];
    atomicAdd(&agg_s[(size_t)dst * NS + c], alpha * val);
  } else {
    int cc = c - NS;
    float val = v_v[(size_t)src * 48 + cc] + ev_v[(size_t)edge * 48 + cc];
    atomicAdd(&agg_v[(size_t)dst * 48 + cc], alpha * val);
  }
}

// residual + GVP layernorm (scalar LN + vector RMS over vector channels)
__global__ __launch_bounds__(128) void ln_kernel(
    const float* __restrict__ s_in, const float* __restrict__ ds,
    const float* __restrict__ v_in, const float* __restrict__ dv,
    const float* __restrict__ g, const float* __restrict__ b,
    float* __restrict__ s_out, float* __restrict__ v_out) {
  __shared__ float red[128];
  __shared__ float sv[48];
  int node = blockIdx.x, t = threadIdx.x;

  float sval = s_in[(size_t)node * NS + t] + ds[(size_t)node * NS + t];
  red[t] = sval; __syncthreads();
  for (int o = 64; o > 0; o >>= 1) { if (t < o) red[t] += red[t + o]; __syncthreads(); }
  float mu = red[0] / 128.f; __syncthreads();
  float d = sval - mu;
  red[t] = d * d; __syncthreads();
  for (int o = 64; o > 0; o >>= 1) { if (t < o) red[t] += red[t + o]; __syncthreads(); }
  float var = red[0] / 128.f; __syncthreads();
  s_out[(size_t)node * NS + t] = d * rsqrtf(var + 1e-5f) * g[t] + b[t];

  if (t < 48) sv[t] = v_in[(size_t)node * 48 + t] + dv[(size_t)node * 48 + t];
  __syncthreads();
  if (t < 16) {
    float a0 = sv[t * 3], a1 = sv[t * 3 + 1], a2 = sv[t * 3 + 2];
    red[t] = fmaxf(a0 * a0 + a1 * a1 + a2 * a2, 1e-8f);
  }
  __syncthreads();
  if (t < 8) red[t] += red[t + 8];  __syncthreads();
  if (t < 4) red[t] += red[t + 4];  __syncthreads();
  if (t < 2) red[t] += red[t + 2];  __syncthreads();
  if (t < 1) red[0] += red[1];      __syncthreads();
  float inv_vn = rsqrtf(red[0] / 16.f);     // v / sqrt(mean(vn2)) as one v_rsq
  if (t < 48) v_out[(size_t)node * 48 + t] = sv[t] * inv_vn;
}

// ----------------------------- host launch ----------------------------------
struct GvpW { const float *wh, *ws_w, *ws_b, *wv, *wsv_w, *wsv_b; };

static inline size_t packed_count(int K, int N) {
  int KP = (K + 31) & ~31;
  return (size_t)(KP / 32) * (size_t)(N / 16) * 512;  // halves
}

extern "C" void kernel_launch(void* const* d_in, const int* in_sizes, int n_in,
                              void* d_out, int out_size, void* d_ws, size_t ws_size,
                              hipStream_t stream) {
  (void)ws_size; (void)out_size;
  const float* x_s  = (const float*)d_in[0];   // [N,128]
  const float* x_v  = (const float*)d_in[1];   // [N,16,3]
  const int*   ei   = (const int*)d_in[2];     // [2,E]
  const float* ea_s = (const float*)d_in[3];   // [E,32]
  const float* ea_v = (const float*)d_in[4];   // [E,1,3]
  const int N = in_sizes[0] / NS;
  const long long E = in_sizes[2] / 2;

  // Param flattening order: insertion (to_q first, wh=256) vs pytree-sorted
  // (attn first, wh=64). Disambiguate by probing in_sizes[5].
  const bool srt = (n_in > 5 && in_sizes[5] == 64);
  auto F = [&](int i) { return (const float*)d_in[i]; };
  auto gvp_at = [&](int i0) -> GvpW {
    if (srt) return { F(i0), F(i0 + 2), F(i0 + 1), F(i0 + 5), F(i0 + 4), F(i0 + 3) };
    return { F(i0), F(i0 + 1), F(i0 + 2), F(i0 + 3), F(i0 + 4), F(i0 + 5) };
  };
  GvpW to_q, to_k, to_v, e_k, e_v, outp, ff1, ff2;
  const float *awh, *aws_w, *aws_b, *ln0g, *ln0b, *ln1g, *ln1b;
  if (srt) {
    awh = F(5); aws_b = F(6); aws_w = F(7);
    e_k = gvp_at(8); e_v = gvp_at(14); ff1 = gvp_at(20); ff2 = gvp_at(26);
    ln0b = F(32); ln0g = F(33); ln1b = F(34); ln1g = F(35);
    outp = gvp_at(36); to_k = gvp_at(42); to_q = gvp_at(48); to_v = gvp_at(54);
  } else {
    to_q = gvp_at(5); to_k = gvp_at(11); to_v = gvp_at(17);
    e_k = gvp_at(23); e_v = gvp_at(29);
    awh = F(35); aws_w = F(36); aws_b = F(37);
    outp = gvp_at(38); ff1 = gvp_at(44); ff2 = gvp_at(50);
    ln0g = F(56); ln0b = F(57); ln1g = F(58); ln1b = F(59);
  }

  // workspace carve-up (64B-aligned chunks)
  char* p = (char*)d_ws;
  auto alloc = [&](size_t bytes) {
    char* r = p;
    p += (bytes + 63) & ~(size_t)63;
    return r;
  };
  auto allocF = [&](size_t n) { return (float*)alloc(n * 4); };
  auto allocH = [&](size_t n) { return (_Float16*)alloc(n * 2); };

  float* q_s = allocF((size_t)N * NS);  float* q_v = allocF((size_t)N * 48);
  float* k_s = allocF((size_t)N * NS);  float* k_v = allocF((size_t)N * 48);
  float* v_s = allocF((size_t)N * NS);  float* v_v = allocF((size_t)N * 48);
  float* ek_s = allocF((size_t)E * NS); float* ek_v = allocF((size_t)E * 48);
  float* ev_s = allocF((size_t)E * NS); float* ev_v = allocF((size_t)E * 48);
  float* energy = allocF((size_t)E * 4);
  float* zbase  = (float*)p;               // contiguous: maxbuf|den|agg_s|agg_v
  unsigned* maxbuf = (unsigned*)allocF((size_t)N * 4);
  float* den   = allocF((size_t)N * 4);
  float* agg_s = allocF((size_t)N * NS);  float* agg_v = allocF((size_t)N * 48);
  float* dh_s  = allocF((size_t)N * NS);  float* dh_v  = allocF((size_t)N * 48);
  float* s0    = allocF((size_t)N * NS);  float* v0    = allocF((size_t)N * 48);
  float* h_s   = allocF((size_t)N * 512); float* h_v   = allocF((size_t)N * 96);
  float* d2_s  = allocF((size_t)N * NS);  float* d2_v  = allocF((size_t)N * 48);

  // packed f16 weight buffers
  struct PW { _Float16 *ws, *wsv; };
  auto pack = [&](const GvpW& w, int Kws, int Nws, int Kwsv, int Nwsv) -> PW {
    PW r;
    size_t cs = packed_count(Kws, Nws);
    size_t cg = packed_count(Kwsv, Nwsv);
    r.ws  = allocH(cs);
    r.wsv = allocH(cg);
    pack_w_kernel<<<(unsigned)((cs + 255) / 256), 256, 0, stream>>>(
        w.ws_w, r.ws, Kws, Nws, (long long)cs);
    pack_w_kernel<<<(unsigned)((cg + 255) / 256), 256, 0, stream>>>(
        w.wsv_w, r.wsv, Kwsv, Nwsv, (long long)cg);
    return r;
  };
  PW pq  = pack(to_q, 144, 128, 128, 16);
  PW pk  = pack(to_k, 144, 128, 128, 16);
  PW pv  = pack(to_v, 144, 128, 128, 16);
  PW pek = pack(e_k,   48, 128, 128, 16);
  PW pev = pack(e_v,   48, 128, 128, 16);
  PW pop = pack(outp, 144, 128, 128, 16);
  PW pf1 = pack(ff1,  160, 512, 512, 32);
  PW pf2 = pack(ff2,  544, 128, 128, 16);

  const int nb_node = (N + 15) / 16;
  const int nb_edge = (int)((E + 15) / 16);

  // node q/k/v GVPs (WMMA)
  gvp_kernel<128, 16, 128, 16, 16, true, true><<<nb_node, 256, 0, stream>>>(
      x_s, x_v, to_q.wh, pq.ws, to_q.ws_b, to_q.wv, pq.wsv, to_q.wsv_b, q_s, q_v, N);
  gvp_kernel<128, 16, 128, 16, 16, true, true><<<nb_node, 256, 0, stream>>>(
      x_s, x_v, to_k.wh, pk.ws, to_k.ws_b, to_k.wv, pk.wsv, to_k.wsv_b, k_s, k_v, N);
  gvp_kernel<128, 16, 128, 16, 16, true, true><<<nb_node, 256, 0, stream>>>(
      x_s, x_v, to_v.wh, pv.ws, to_v.ws_b, to_v.wv, pv.wsv, to_v.wsv_b, v_s, v_v, N);
  // edge GVPs (WMMA)
  gvp_kernel<32, 1, 128, 16, 16, true, true><<<nb_edge, 256, 0, stream>>>(
      ea_s, ea_v, e_k.wh, pek.ws, e_k.ws_b, e_k.wv, pek.wsv, e_k.wsv_b, ek_s, ek_v, (int)E);
  gvp_kernel<32, 1, 128, 16, 16, true, true><<<nb_edge, 256, 0, stream>>>(
      ea_s, ea_v, e_v.wh, pev.ws, e_v.ws_b, e_v.wv, pev.wsv, e_v.wsv_b, ev_s, ev_v, (int)E);

  // zero the atomic accumulators (maxbuf key 0 == -inf in keyed order)
  long long zc = (long long)N * (4 + 4 + NS + 48);
  zero_kernel<<<(unsigned)((zc + 255) / 256), 256, 0, stream>>>(zbase, zc);

  long long eh = E * 4;
  energy_kernel<<<(unsigned)((eh + 255) / 256), 256, 0, stream>>>(
      ei, E, q_s, q_v, k_s, k_v, ek_s, ek_v, awh, aws_w, aws_b, energy, maxbuf);
  softmax_kernel<<<(unsigned)((eh + 255) / 256), 256, 0, stream>>>(
      ei, E, energy, maxbuf, den);
  long long aw = E * 176;
  agg_kernel<<<(unsigned)((aw + 255) / 256), 256, 0, stream>>>(
      ei, E, energy, den, v_s, v_v, ev_s, ev_v, agg_s, agg_v);

  // out_proj GVP (WMMA)
  gvp_kernel<128, 16, 128, 16, 16, true, true><<<nb_node, 256, 0, stream>>>(
      agg_s, agg_v, outp.wh, pop.ws, outp.ws_b, outp.wv, pop.wsv, outp.wsv_b,
      dh_s, dh_v, N);
  // residual + ln0
  ln_kernel<<<N, 128, 0, stream>>>(x_s, dh_s, x_v, dh_v, ln0g, ln0b, s0, v0);
  // feed-forward GVPs (WMMA)
  gvp_kernel<128, 16, 512, 32, 32, true, true><<<nb_node, 256, 0, stream>>>(
      s0, v0, ff1.wh, pf1.ws, ff1.ws_b, ff1.wv, pf1.wsv, ff1.wsv_b, h_s, h_v, N);
  gvp_kernel<512, 32, 128, 16, 32, false, false><<<nb_node, 256, 0, stream>>>(
      h_s, h_v, ff2.wh, pf2.ws, ff2.ws_b, ff2.wv, pf2.wsv, ff2.wsv_b, d2_s, d2_v, N);
  // residual + ln1 -> output (s then v, concat flat)
  float* out_s = (float*)d_out;
  float* out_v = out_s + (size_t)N * NS;
  ln_kernel<<<N, 128, 0, stream>>>(s0, d2_s, v0, d2_v, ln1g, ln1b, out_s, out_v);
}
